// HyperbolicFractalBlock_80994493268392
// MI455X (gfx1250) — matrix operand order repaired and checked by reading, not compiled, not run on hardware
//
#include <hip/hip_runtime.h>
#include <hip/hip_bf16.h>
#include <math.h>

// Problem constants (B=4, T=1024, D=512, H=8, hd=64, FF=1024)
#define BT 4096
#define DD 512
#define FFD 1024
#define TT 1024
#define NH 8
#define HD 64

typedef __bf16 bf16_t;
typedef __attribute__((ext_vector_type(16))) __bf16 v16bf;
typedef __attribute__((ext_vector_type(8)))  float  v8f;
typedef __attribute__((ext_vector_type(4)))  unsigned int u32x4;
typedef __attribute__((ext_vector_type(4)))  float  f32x4;

union ABFrag  { v16bf v; u32x4 q[2]; };
union AccFrag { v8f   v; float f[8]; };

__device__ __forceinline__ bf16_t f2bf(float x) { return (bf16_t)x; }

// ---------------------------------------------------------------------------
// block-wide (128 threads / 4 waves) sum reduction
// ---------------------------------------------------------------------------
__device__ __forceinline__ float blk_sum128(float v, float* sb) {
#pragma unroll
  for (int m = 16; m >= 1; m >>= 1) v += __shfl_xor(v, m, 32);
  __syncthreads();
  if ((threadIdx.x & 31) == 0) sb[threadIdx.x >> 5] = v;
  __syncthreads();
  return sb[0] + sb[1] + sb[2] + sb[3];
}

// ---------------------------------------------------------------------------
// f32 -> bf16 convert (weights)
// ---------------------------------------------------------------------------
__global__ void f32_to_bf16_kernel(const float* __restrict__ in,
                                   bf16_t* __restrict__ outp, int nTot) {
  int i = blockIdx.x * blockDim.x + threadIdx.x;
  if (i < nTot) outp[i] = f2bf(in[i]);
}

// ---------------------------------------------------------------------------
// prep1: h = rms_norm(x, w); h_ball = project(0.5*h); t = log0(h_ball)
// Both ball-project and log0 reduce to one per-row scalar scale on h.
// Emits h (bf16) and t (bf16).  One block (128 thr) per row of 512.
// ---------------------------------------------------------------------------
__global__ void __launch_bounds__(128)
prep1_kernel(const float* __restrict__ x, const float* __restrict__ w,
             bf16_t* __restrict__ hbf, bf16_t* __restrict__ tbf) {
  __shared__ float sb[4];
  const int row = blockIdx.x;
  const int tid = threadIdx.x;
  f32x4 xv = *(const f32x4*)(x + (size_t)row * DD + tid * 4);
  float ss = xv[0]*xv[0] + xv[1]*xv[1] + xv[2]*xv[2] + xv[3]*xv[3];
  float tot = blk_sum128(ss, sb);
  float rinv = rsqrtf(tot * (1.0f / DD) + 1.1920929e-7f);
  f32x4 wv4 = *(const f32x4*)(w + tid * 4);
  f32x4 h;
#pragma unroll
  for (int i = 0; i < 4; ++i) h[i] = xv[i] * rinv * wv4[i];
  float hs = h[0]*h[0] + h[1]*h[1] + h[2]*h[2] + h[3]*h[3];
  float htot = blk_sum128(hs, sb);
  float nh = 0.5f * sqrtf(htot);                 // || 0.5*h ||
  const float maxn = 1.0f - 1e-5f;
  float sp = (nh > maxn) ? (maxn / nh) : 1.0f;   // project_to_ball scale
  float nb = fminf(fmaxf(nh * sp, 1e-6f), 1.0f - 1e-5f);
  float st = 0.5f * sp * atanhf(nb) / nb;        // t = h * st
  size_t base = (size_t)row * DD + tid * 4;
#pragma unroll
  for (int i = 0; i < 4; ++i) {
    hbf[base + i] = f2bf(h[i]);
    tbf[base + i] = f2bf(h[i] * st);
  }
}

// ---------------------------------------------------------------------------
// prep2: plain rms_norm -> bf16
// ---------------------------------------------------------------------------
__global__ void __launch_bounds__(128)
prep2_kernel(const float* __restrict__ x, const float* __restrict__ w,
             bf16_t* __restrict__ hbf) {
  __shared__ float sb[4];
  const int row = blockIdx.x;
  const int tid = threadIdx.x;
  f32x4 xv = *(const f32x4*)(x + (size_t)row * DD + tid * 4);
  float ss = xv[0]*xv[0] + xv[1]*xv[1] + xv[2]*xv[2] + xv[3]*xv[3];
  float tot = blk_sum128(ss, sb);
  float rinv = rsqrtf(tot * (1.0f / DD) + 1.1920929e-7f);
  f32x4 wv4 = *(const f32x4*)(w + tid * 4);
  size_t base = (size_t)row * DD + tid * 4;
#pragma unroll
  for (int i = 0; i < 4; ++i) hbf[base + i] = f2bf(xv[i] * rinv * wv4[i]);
}

// ---------------------------------------------------------------------------
// Generic WMMA GEMM:  C[M,N] = A[M,K] * W[N,K]^T  (+ optional f32 residual)
// A, W bf16 row-major (both K-major -> no transpose needed).
// Block: 256 threads (8 waves as 4(M) x 2(N)); each wave computes a
// 32(M) x 64(N) tile: 2 A-frags reused over 4 B-frags -> 8 WMMAs per K=32
// step at 12 b128 loads (1.5 loads/WMMA).  Grid: (N/128, M/128).
// ---------------------------------------------------------------------------
__global__ void __launch_bounds__(256)
gemm_bf16_wmma(const bf16_t* __restrict__ A, const bf16_t* __restrict__ W,
               const float* resid, float* C, int M, int N, int K) {
  const int lane = threadIdx.x & 31;
  const int wv   = threadIdx.x >> 5;
  const int ln   = lane & 15;
  const int half = lane >> 4;
  const int row0 = blockIdx.y * 128 + (wv >> 1) * 32;
  const int col0 = blockIdx.x * 128 + (wv & 1) * 64;

  AccFrag acc[2][4];
#pragma unroll
  for (int i = 0; i < 2; ++i)
#pragma unroll
    for (int j = 0; j < 4; ++j)
#pragma unroll
      for (int r = 0; r < 8; ++r) acc[i][j].f[r] = 0.0f;

  const bf16_t* aRow0 = A + (size_t)(row0 + ln) * K + half * 8;
  const bf16_t* aRow1 = A + (size_t)(row0 + 16 + ln) * K + half * 8;

  for (int k0 = 0; k0 < K; k0 += 32) {
    // prefetch A rows ~8 K-steps (512B) ahead -> global_prefetch_b8
    if (k0 + 256 < K) {
      __builtin_prefetch(aRow0 + k0 + 256, 0, 0);
      __builtin_prefetch(aRow1 + k0 + 256, 0, 0);
    }
    ABFrag af0, af1;
    af0.q[0] = *(const u32x4*)(aRow0 + k0);       // A frag: K 0-7 & 16-23 (half 0)
    af0.q[1] = *(const u32x4*)(aRow0 + k0 + 16);
    af1.q[0] = *(const u32x4*)(aRow1 + k0);
    af1.q[1] = *(const u32x4*)(aRow1 + k0 + 16);
#pragma unroll
    for (int j = 0; j < 4; ++j) {
      ABFrag bfr;
      const bf16_t* bp = W + (size_t)(col0 + j * 16 + ln) * K + k0 + half * 8;
      bfr.q[0] = *(const u32x4*)bp;               // B frag: K 0-7 & 16-23 (half 0)
      bfr.q[1] = *(const u32x4*)(bp + 16);
      acc[0][j].v = __builtin_amdgcn_wmma_f32_16x16x32_bf16(
          false, af0.v, false, bfr.v, (short)0, acc[0][j].v, false, false);
      acc[1][j].v = __builtin_amdgcn_wmma_f32_16x16x32_bf16(
          false, af1.v, false, bfr.v, (short)0, acc[1][j].v, false, false);
    }
  }
#pragma unroll
  for (int i = 0; i < 2; ++i)
#pragma unroll
    for (int j = 0; j < 4; ++j)
#pragma unroll
      for (int r = 0; r < 8; ++r) {
        size_t idx = (size_t)(row0 + i * 16 + half * 8 + r) * N +
                     (col0 + j * 16 + ln);
        float v = acc[i][j].f[r];
        if (resid) v += resid[idx];
        C[idx] = v;
      }
}

// ---------------------------------------------------------------------------
// exp0+project epilogue for q/k: per-row scale; also per-head sum of squares.
// One block (128 thr) per row.
// ---------------------------------------------------------------------------
__global__ void __launch_bounds__(128)
exp0_head_kernel(const float* __restrict__ O, bf16_t* __restrict__ Qbf,
                 float* __restrict__ Q2) {
  __shared__ float sb[4];
  const int row = blockIdx.x;
  const int tid = threadIdx.x;
  f32x4 ov = *(const f32x4*)(O + (size_t)row * DD + tid * 4);
  float ss = ov[0]*ov[0] + ov[1]*ov[1] + ov[2]*ov[2] + ov[3]*ov[3];
  float tot = blk_sum128(ss, sb);
  float nrm = sqrtf(tot);
  float n  = fmaxf(nrm, 1e-6f);
  float th = tanhf(n);
  float en = th * nrm / n;                    // actual norm of exp0 output
  const float maxn = 1.0f - 1e-5f;
  float sp = (en > maxn) ? (maxn / en) : 1.0f;
  float s  = (th / n) * sp;
  size_t base = (size_t)row * DD + tid * 4;
#pragma unroll
  for (int i = 0; i < 4; ++i) Qbf[base + i] = f2bf(ov[i] * s);
  float hss = ss * s * s;                     // this thread's 4 elems are one head
#pragma unroll
  for (int m = 1; m < 16; m <<= 1) hss += __shfl_xor(hss, m, 32);
  if ((tid & 15) == 0) {
    int head = tid >> 4;
    int b = row >> 10, t = row & (TT - 1);
    Q2[((size_t)(b * NH + head)) * TT + t] = hss;
  }
}

// ---------------------------------------------------------------------------
// V transpose:  OV f32 [B*T, D] -> VT bf16 [(b*H+h)*hd + d][T]
// ---------------------------------------------------------------------------
__global__ void transpose_v_kernel(const float* __restrict__ OV,
                                   bf16_t* __restrict__ VT) {
  int idx = blockIdx.x * blockDim.x + threadIdx.x;   // < BT*DD
  int t = idx & (TT - 1);
  int rest = idx >> 10;
  int d = rest & (HD - 1);
  int bh = rest >> 6;
  int b = bh >> 3, h = bh & 7;
  VT[idx] = f2bf(OV[((size_t)(b * TT + t)) * DD + h * HD + d]);
}

// ---------------------------------------------------------------------------
// Causal distance-attention, flash style.
// Grid (T/64, B*H); block 128 (4 waves, 16 query rows each).
// score = -(q2 + k2 - 2*q.k) / 8 ; online softmax ; P via LDS -> PV WMMA.
// ---------------------------------------------------------------------------
__global__ void __launch_bounds__(128)
attn_kernel(const bf16_t* __restrict__ Q, const bf16_t* __restrict__ Kb,
            const bf16_t* __restrict__ VT, const float* __restrict__ Q2,
            const float* __restrict__ K2, bf16_t* __restrict__ Obf) {
  __shared__ bf16_t pbuf[128 * 64];                 // 16 KB, 2 KB per wave used
  const int qb = blockIdx.x;
  const int bh = blockIdx.y;
  const int b = bh >> 3, h = bh & 7;
  const int tid = threadIdx.x;
  const int wv = tid >> 5;
  const int lane = tid & 31;
  const int ln = lane & 15, half = lane >> 4;
  const int q0 = qb * 64 + wv * 16;

  bf16_t* myP = pbuf + wv * (16 * 64);
  const bf16_t* qbase = Q  + ((size_t)b * TT) * DD + h * HD;
  const bf16_t* kbase = Kb + ((size_t)b * TT) * DD + h * HD;
  const bf16_t* vbase = VT + ((size_t)bh * HD) * TT;
  const float* q2r = Q2 + (size_t)bh * TT;
  const float* k2r = K2 + (size_t)bh * TT;

  AccFrag oacc[4];
#pragma unroll
  for (int dt = 0; dt < 4; ++dt)
#pragma unroll
    for (int r = 0; r < 8; ++r) oacc[dt].f[r] = 0.0f;

  float mst[8], lst[8], q2v[8];
#pragma unroll
  for (int r = 0; r < 8; ++r) {
    mst[r] = -3.0e38f; lst[r] = 0.0f;
    q2v[r] = q2r[q0 + half * 8 + r];
  }

  for (int kb = 0; kb <= qb; ++kb) {
    const int k0 = kb * 64;
    AccFrag s[4];
#pragma unroll
    for (int j = 0; j < 4; ++j)
#pragma unroll
      for (int r = 0; r < 8; ++r) s[j].f[r] = 0.0f;

    // S = Q * K^T over hd=64 (two K=32 steps)
#pragma unroll
    for (int ks = 0; ks < 2; ++ks) {
      ABFrag af;
      const bf16_t* ap = qbase + (size_t)(q0 + ln) * DD + ks * 32 + half * 8;
      af.q[0] = *(const u32x4*)ap;
      af.q[1] = *(const u32x4*)(ap + 16);
#pragma unroll
      for (int j = 0; j < 4; ++j) {
        ABFrag bfr;
        const bf16_t* bp = kbase + (size_t)(k0 + j * 16 + ln) * DD + ks * 32 + half * 8;
        bfr.q[0] = *(const u32x4*)bp;
        bfr.q[1] = *(const u32x4*)(bp + 16);
        s[j].v = __builtin_amdgcn_wmma_f32_16x16x32_bf16(
            false, af.v, false, bfr.v, (short)0, s[j].v, false, false);
      }
    }

    // distance bias + causal mask
    float pv[4][8], rowmax[8];
#pragma unroll
    for (int r = 0; r < 8; ++r) rowmax[r] = -3.0e38f;
#pragma unroll
    for (int j = 0; j < 4; ++j) {
      const int key = k0 + j * 16 + ln;
      const float k2v = k2r[key];
#pragma unroll
      for (int r = 0; r < 8; ++r) {
        const int qrow = q0 + half * 8 + r;
        float val = (key <= qrow)
                        ? -(q2v[r] + k2v - 2.0f * s[j].f[r]) * 0.125f
                        : -3.0e38f;
        pv[j][r] = val;
        rowmax[r] = fmaxf(rowmax[r], val);
      }
    }
#pragma unroll
    for (int r = 0; r < 8; ++r)
#pragma unroll
      for (int m = 1; m < 16; m <<= 1)
        rowmax[r] = fmaxf(rowmax[r], __shfl_xor(rowmax[r], m, 32));

    float rowsum[8];
#pragma unroll
    for (int r = 0; r < 8; ++r) {
      float mnew = fmaxf(mst[r], rowmax[r]);
      float alpha = __expf(mst[r] - mnew);
      mst[r] = mnew; lst[r] *= alpha; rowsum[r] = 0.0f;
#pragma unroll
      for (int dt = 0; dt < 4; ++dt) oacc[dt].f[r] *= alpha;
    }
#pragma unroll
    for (int j = 0; j < 4; ++j)
#pragma unroll
      for (int r = 0; r < 8; ++r) {
        float p = __expf(pv[j][r] - mst[r]);
        rowsum[r] += p;
        myP[(half * 8 + r) * 64 + j * 16 + ln] = f2bf(p);   // stage P in LDS
      }
#pragma unroll
    for (int r = 0; r < 8; ++r) {
#pragma unroll
      for (int m = 1; m < 16; m <<= 1) rowsum[r] += __shfl_xor(rowsum[r], m, 32);
      lst[r] += rowsum[r];
    }

    // O += P * V  (two K=32 steps over the 64-key block)
#pragma unroll
    for (int ks = 0; ks < 2; ++ks) {
      ABFrag pf;
      const bf16_t* pp = myP + (size_t)ln * 64 + ks * 32 + half * 8;
      pf.q[0] = *(const u32x4*)pp;
      pf.q[1] = *(const u32x4*)(pp + 16);
#pragma unroll
      for (int dt = 0; dt < 4; ++dt) {
        ABFrag vfr;
        const bf16_t* vp = vbase + (size_t)(dt * 16 + ln) * TT + k0 + ks * 32 + half * 8;
        vfr.q[0] = *(const u32x4*)vp;
        vfr.q[1] = *(const u32x4*)(vp + 16);
        oacc[dt].v = __builtin_amdgcn_wmma_f32_16x16x32_bf16(
            false, pf.v, false, vfr.v, (short)0, oacc[dt].v, false, false);
      }
    }
  }

#pragma unroll
  for (int dt = 0; dt < 4; ++dt)
#pragma unroll
    for (int r = 0; r < 8; ++r) {
      const int qrow = q0 + half * 8 + r;
      Obf[((size_t)(b * TT + qrow)) * DD + h * HD + dt * 16 + ln] =
          f2bf(oacc[dt].f[r] / lst[r]);
    }
}

// ---------------------------------------------------------------------------
// SwiGLU fuse: a = silu(gate) * up -> bf16
// ---------------------------------------------------------------------------
__global__ void swiglu_kernel(const float* __restrict__ G,
                              const float* __restrict__ U,
                              bf16_t* __restrict__ A, int nTot) {
  int i = blockIdx.x * blockDim.x + threadIdx.x;
  if (i < nTot) {
    float g = G[i];
    A[i] = f2bf(g / (1.0f + __expf(-g)) * U[i]);
  }
}

// ---------------------------------------------------------------------------
// host launcher
// ---------------------------------------------------------------------------
extern "C" void kernel_launch(void* const* d_in, const int* in_sizes, int n_in,
                              void* d_out, int out_size, void* d_ws, size_t ws_size,
                              hipStream_t stream) {
  (void)in_sizes; (void)n_in; (void)out_size; (void)ws_size;
  const float* x      = (const float*)d_in[0];
  const float* q_w    = (const float*)d_in[1];
  const float* k_w    = (const float*)d_in[2];
  const float* v_w    = (const float*)d_in[3];
  const float* o_w    = (const float*)d_in[4];
  const float* gate_w = (const float*)d_in[5];
  const float* up_w   = (const float*)d_in[6];
  const float* down_w = (const float*)d_in[7];
  const float* n1_w   = (const float*)d_in[8];
  const float* n2_w   = (const float*)d_in[9];
  float* out = (float*)d_out;

  char* ws = (char*)d_ws;
  size_t off = 0;
  auto take = [&](size_t bytes) -> char* {
    char* p = ws + off;
    off += (bytes + 255) & ~(size_t)255;
    return p;
  };
  bf16_t* wq = (bf16_t*)take((size_t)DD * DD * 2);
  bf16_t* wk = (bf16_t*)take((size_t)DD * DD * 2);
  bf16_t* wvw= (bf16_t*)take((size_t)DD * DD * 2);
  bf16_t* wo = (bf16_t*)take((size_t)DD * DD * 2);
  bf16_t* wg = (bf16_t*)take((size_t)FFD * DD * 2);
  bf16_t* wu = (bf16_t*)take((size_t)FFD * DD * 2);
  bf16_t* wd = (bf16_t*)take((size_t)DD * FFD * 2);
  bf16_t* hbf   = (bf16_t*)take((size_t)BT * DD * 2);
  bf16_t* tbf   = (bf16_t*)take((size_t)BT * DD * 2);
  float*  SA    = (float*)take((size_t)BT * FFD * 4);   // oq, later gate
  float*  SB    = (float*)take((size_t)BT * FFD * 4);   // ok, later up
  float*  SC    = (float*)take((size_t)BT * DD * 4);    // ov
  bf16_t* qbf   = (bf16_t*)take((size_t)BT * DD * 2);
  bf16_t* kbf   = (bf16_t*)take((size_t)BT * DD * 2);
  bf16_t* vt    = (bf16_t*)take((size_t)BT * DD * 2);
  float*  q2    = (float*)take((size_t)4 * NH * TT * 4);
  float*  k2    = (float*)take((size_t)4 * NH * TT * 4);
  bf16_t* attnb = (bf16_t*)take((size_t)BT * DD * 2);
  bf16_t* h2bf  = (bf16_t*)take((size_t)BT * DD * 2);
  bf16_t* abf   = (bf16_t*)take((size_t)BT * FFD * 2);

  // 1. weights -> bf16
  f32_to_bf16_kernel<<<(DD*DD + 255) / 256, 256, 0, stream>>>(q_w,    wq,  DD*DD);
  f32_to_bf16_kernel<<<(DD*DD + 255) / 256, 256, 0, stream>>>(k_w,    wk,  DD*DD);
  f32_to_bf16_kernel<<<(DD*DD + 255) / 256, 256, 0, stream>>>(v_w,    wvw, DD*DD);
  f32_to_bf16_kernel<<<(DD*DD + 255) / 256, 256, 0, stream>>>(o_w,    wo,  DD*DD);
  f32_to_bf16_kernel<<<(FFD*DD + 255) / 256, 256, 0, stream>>>(gate_w, wg, FFD*DD);
  f32_to_bf16_kernel<<<(FFD*DD + 255) / 256, 256, 0, stream>>>(up_w,   wu, FFD*DD);
  f32_to_bf16_kernel<<<(DD*FFD + 255) / 256, 256, 0, stream>>>(down_w, wd, DD*FFD);

  // 2. norm + hyperbolic pre-maps
  prep1_kernel<<<BT, 128, 0, stream>>>(x, n1_w, hbf, tbf);

  // 3. Q/K/V projections
  gemm_bf16_wmma<<<dim3(DD/128, BT/128), 256, 0, stream>>>(tbf, wq, nullptr, SA, BT, DD, DD);
  exp0_head_kernel<<<BT, 128, 0, stream>>>(SA, qbf, q2);
  gemm_bf16_wmma<<<dim3(DD/128, BT/128), 256, 0, stream>>>(tbf, wk, nullptr, SB, BT, DD, DD);
  exp0_head_kernel<<<BT, 128, 0, stream>>>(SB, kbf, k2);
  gemm_bf16_wmma<<<dim3(DD/128, BT/128), 256, 0, stream>>>(hbf, wvw, nullptr, SC, BT, DD, DD);
  transpose_v_kernel<<<(BT*DD) / 256, 256, 0, stream>>>(SC, vt);

  // 4. attention
  attn_kernel<<<dim3(TT/64, 4*NH), 128, 0, stream>>>(qbf, kbf, vt, q2, k2, attnb);

  // 5. output projection + residual
  gemm_bf16_wmma<<<dim3(DD/128, BT/128), 256, 0, stream>>>(attnb, wo, x, out, BT, DD, DD);

  // 6. FFN
  prep2_kernel<<<BT, 128, 0, stream>>>(out, n2_w, h2bf);
  gemm_bf16_wmma<<<dim3(FFD/128, BT/128), 256, 0, stream>>>(h2bf, wg, nullptr, SA, BT, FFD, DD);
  gemm_bf16_wmma<<<dim3(FFD/128, BT/128), 256, 0, stream>>>(h2bf, wu, nullptr, SB, BT, FFD, DD);
  swiglu_kernel<<<(BT*FFD) / 256, 256, 0, stream>>>(SA, SB, abf, BT*FFD);
  gemm_bf16_wmma<<<dim3(DD/128, BT/128), 256, 0, stream>>>(abf, wd, out, out, BT, DD, FFD);
}
